// CustomTacotron2Decoder_16260746182910
// MI455X (gfx1250) — matrix operand order repaired and checked by reading, not compiled, not run on hardware
//
#include <hip/hip_runtime.h>
#include <cstdint>
#include <cstddef>

// ---------------------------------------------------------------------------
// CDNA5 (gfx1250) fused Tacotron2 LSTM decoder.
// B=256, T=1024, HID=128, IN=128 (speech 64 | word 64), OUT=64.
// 16 WGs x 256 threads (8 wave32); each WG owns 16 batch rows.
// Per step: gates(16x512) = [h|speech|word](16x256) @ Bpacked(256x512) via
// v_wmma_f32_16x16x32_bf16 (weights resident in VGPRs), lane-local LSTM cell
// update, h -> LDS (2 barriers/step), fc(16x64) on waves 0-3, store y.
// Word tile for t+1 is global-loaded at iteration top to overlap the GEMM.
// ---------------------------------------------------------------------------

typedef __attribute__((ext_vector_type(16))) __bf16 v16bf;
typedef __attribute__((ext_vector_type(8)))  float  v8f;

#define T_LEN 1024
#define H_STRIDE 136   // halves per row of h_lds  (272B, 16B-aligned, bank-spread)
#define W_STRIDE 72    // halves per row of w_lds  (144B, 16B-aligned, bank-spread)

#define GATES_B_BYTES (32 * 8 * 32 * 32)   // 32 Ntiles x 8 kblocks x 32 lanes x 32B
#define FC_B_BYTES    (4 * 4 * 32 * 32)    // 4 Ntiles x 4 kblocks x 32 lanes x 32B

__device__ __forceinline__ unsigned short f2bf(float f) {
  union { float f; unsigned u; } x; x.f = f;
  unsigned r = x.u + 0x7FFFu + ((x.u >> 16) & 1u);   // round-to-nearest-even
  return (unsigned short)(r >> 16);
}

// gfx1250 has V_TANH_F32; use it if the builtin is declared, else expf fallback.
#if defined(__has_builtin)
#if __has_builtin(__builtin_amdgcn_tanhf)
#define FAST_TANH(x) __builtin_amdgcn_tanhf(x)
#elif __has_builtin(__builtin_amdgcn_tanh_f32)
#define FAST_TANH(x) __builtin_amdgcn_tanh_f32(x)
#endif
#endif
#ifndef FAST_TANH
__device__ __forceinline__ float fast_tanh_fallback(float x) {
  return 1.0f - 2.0f / (__expf(2.0f * x) + 1.0f);
}
#define FAST_TANH(x) fast_tanh_fallback(x)
#endif

__device__ __forceinline__ float fsig(float x) {
  return 0.5f + 0.5f * FAST_TANH(0.5f * x);
}

union FragA { v16bf v; uint4 q[2]; };

// A-matrix (16x32 bf16) fragment gather per ISA layout:
// lane l: row M = l%16; lanes 0-15 hold K = {0..7, 16..23}, lanes 16-31 K+8.
__device__ __forceinline__ v16bf lds_gatherA(const unsigned short* arr, int strideH,
                                             int lrow, int lhalf, int kbaseH) {
  FragA f;
  const unsigned short* p = arr + lrow * strideH + kbaseH + 8 * lhalf;
  f.q[0] = *(const uint4*)(p);        // K = kbase + 8*half + 0..7
  f.q[1] = *(const uint4*)(p + 16);   // K = kbase + 16 + 8*half + 0..7
  return f.v;
}

__device__ __forceinline__ v8f splat8(float x) {
  v8f r;
#pragma unroll
  for (int i = 0; i < 8; ++i) r[i] = x;
  return r;
}

// ---------------------------------------------------------------------------
// Prep: pack W into WMMA B-fragment layout (bf16) + combined gate bias.
// B-matrix (32x16 bf16) layout: lane l holds N=l%16; lanes 0-15 K={0..7,16..23},
// lanes 16-31 K={8..15,24..31} (matches A-fragment K order).
// Combined K axis: k<128 -> W_hh[n][k]; k>=128 -> W_ih[n][k-128]
// (combined input = [speech(0..63) | word(64..127)]).
// ---------------------------------------------------------------------------
__global__ void tacotron_prep_kernel(const float* __restrict__ W_ih,
                                     const float* __restrict__ W_hh,
                                     const float* __restrict__ b_ih,
                                     const float* __restrict__ b_hh,
                                     const float* __restrict__ W_fc,
                                     unsigned short* __restrict__ gatesB,
                                     unsigned short* __restrict__ fcB,
                                     float* __restrict__ biasg) {
  const int idx = blockIdx.x * blockDim.x + threadIdx.x;
  const int NG = 32 * 8 * 32;  // gate lane-chunks
  if (idx < NG) {
    const int lane = idx & 31;
    const int kb   = (idx >> 5) & 7;
    const int tile = idx >> 8;
    const int n    = 16 * tile + (lane & 15);
    const int lhalf = lane >> 4;
    unsigned short* dst = gatesB + idx * 16;
#pragma unroll
    for (int j = 0; j < 16; ++j) {
      const int k = 32 * kb + 8 * lhalf + (j & 7) + 16 * (j >> 3);
      const float v = (k < 128) ? W_hh[n * 128 + k] : W_ih[n * 128 + (k - 128)];
      dst[j] = f2bf(v);
    }
  } else if (idx < NG + 512) {
    const int i2   = idx - NG;
    const int lane = i2 & 31;
    const int kb   = (i2 >> 5) & 3;
    const int tile = i2 >> 7;
    const int n    = 16 * tile + (lane & 15);
    const int lhalf = lane >> 4;
    unsigned short* dst = fcB + i2 * 16;
#pragma unroll
    for (int j = 0; j < 16; ++j) {
      const int k = 32 * kb + 8 * lhalf + (j & 7) + 16 * (j >> 3);
      dst[j] = f2bf(W_fc[n * 128 + k]);
    }
  } else if (idx < NG + 512 + 512) {
    const int g = idx - NG - 512;
    biasg[g] = b_ih[g] + b_hh[g];
  }
}

// ---------------------------------------------------------------------------
// Fused LSTM scan kernel.
// ---------------------------------------------------------------------------
__global__ __launch_bounds__(256, 1)
void tacotron_lstm_scan(const float* __restrict__ speech,
                        const float* __restrict__ word,
                        const v16bf* __restrict__ gatesB,
                        const v16bf* __restrict__ fcB,
                        const float* __restrict__ biasg,
                        const float* __restrict__ b_fc,
                        float* __restrict__ out) {
  __shared__ __align__(16) unsigned short h_lds[16 * H_STRIDE];
  __shared__ __align__(16) unsigned short w_lds[2][16 * W_STRIDE];

  const int tid   = threadIdx.x;
  const int wave  = tid >> 5;
  const int lane  = tid & 31;
  const int lhalf = lane >> 4;
  const int lrow  = lane & 15;
  const int b0    = blockIdx.x * 16;

  // cooperative-loader coordinates (16 rows x 64 cols, 4 cols per thread)
  const int ld_row = (tid * 4) >> 6;
  const int ld_col = (tid * 4) & 63;

  // ---- stage speech rows (bf16) into w_lds[0], gather speech A-fragments ----
  {
    const float4 s4 = *(const float4*)(speech + (size_t)(b0 + ld_row) * 64 + ld_col);
    unsigned short* d = &w_lds[0][ld_row * W_STRIDE + ld_col];
    d[0] = f2bf(s4.x); d[1] = f2bf(s4.y); d[2] = f2bf(s4.z); d[3] = f2bf(s4.w);
  }
  __syncthreads();
  v16bf sA[2];
  sA[0] = lds_gatherA(w_lds[0], W_STRIDE, lrow, lhalf, 0);
  sA[1] = lds_gatherA(w_lds[0], W_STRIDE, lrow, lhalf, 32);
  __syncthreads();

  // ---- zero h state, load word tile for t=0 into w_lds[0] ----
  for (int i = tid; i < 16 * H_STRIDE; i += 256) h_lds[i] = 0;
  {
    const float4 s4 = *(const float4*)(word + ((size_t)(b0 + ld_row) * T_LEN + 0) * 64 + ld_col);
    unsigned short* d = &w_lds[0][ld_row * W_STRIDE + ld_col];
    d[0] = f2bf(s4.x); d[1] = f2bf(s4.y); d[2] = f2bf(s4.z); d[3] = f2bf(s4.w);
  }
  __syncthreads();

  // ---- hoist this wave's gate B-fragments into registers (tiles w, w+8, w+16, w+24) ----
  v16bf Bg[4][8];
#pragma unroll
  for (int j = 0; j < 4; ++j) {
    const int tile = wave + 8 * j;
#pragma unroll
    for (int kb = 0; kb < 8; ++kb)
      Bg[j][kb] = gatesB[(tile * 8 + kb) * 32 + lane];
  }
  // fc B-fragments resident in registers too (used by waves 0-3)
  v16bf Bf[4];
  if (wave < 4) {
#pragma unroll
    for (int kb = 0; kb < 4; ++kb)
      Bf[kb] = fcB[(wave * 4 + kb) * 32 + lane];
  }

  // bias accumulators hoisted: first WMMA of each chain reads these as C directly
  v8f bias_acc[4];
#pragma unroll
  for (int j = 0; j < 4; ++j)
    bias_acc[j] = splat8(biasg[16 * (wave + 8 * j) + lrow]);
  const v8f bfc_acc = splat8((wave < 4) ? b_fc[16 * wave + lrow] : 0.0f);

  float c_reg[8];
#pragma unroll
  for (int r = 0; r < 8; ++r) c_reg[r] = 0.0f;

  // ---- sequential scan ----
  for (int t = 0; t < T_LEN; ++t) {
    const int cur = t & 1;
    const bool havew = (t + 1 < T_LEN);

    // issue the word(t+1) global load NOW so it overlaps the gate GEMM
    float4 wreg;
    if (havew)
      wreg = *(const float4*)(word + ((size_t)(b0 + ld_row) * T_LEN + (t + 1)) * 64 + ld_col);

    // gates(16x512) += A(16x256) x B(256x512); K-blocks: h(0..3) speech(4..5) word(6..7)
    v8f acc[4];
    {
      const v16bf A0 = lds_gatherA(h_lds, H_STRIDE, lrow, lhalf, 0);
#pragma unroll
      for (int j = 0; j < 4; ++j)
        acc[j] = __builtin_amdgcn_wmma_f32_16x16x32_bf16(
            false, A0, false, Bg[j][0], (short)0, bias_acc[j], false, false);
    }
#pragma unroll
    for (int kb = 1; kb < 8; ++kb) {
      v16bf A;
      if (kb < 4)      A = lds_gatherA(h_lds, H_STRIDE, lrow, lhalf, 32 * kb);
      else if (kb < 6) A = sA[kb - 4];
      else             A = lds_gatherA(w_lds[cur], W_STRIDE, lrow, lhalf, 32 * (kb - 6));
#pragma unroll
      for (int j = 0; j < 4; ++j)
        acc[j] = __builtin_amdgcn_wmma_f32_16x16x32_bf16(
            false, A, false, Bg[j][kb], (short)0, acc[j], false, false);
    }

    // lane-local LSTM cell: lane owns hidden unit col=16*wave+lrow, rows r+8*half
    float hnew[8];
#pragma unroll
    for (int r = 0; r < 8; ++r) {
      const float iv = fsig(acc[0][r]);
      const float fv = fsig(acc[1][r]);
      const float gv = FAST_TANH(acc[2][r]);
      const float ov = fsig(acc[3][r]);
      const float c  = fv * c_reg[r] + iv * gv;
      c_reg[r] = c;
      hnew[r]  = ov * FAST_TANH(c);
    }

    __syncthreads();  // all h_lds / w_lds[cur] reads for step t complete

    // publish h_t (bf16); stage the already-loaded word(t+1) tile
    {
      const int col = 16 * wave + lrow;
#pragma unroll
      for (int r = 0; r < 8; ++r)
        h_lds[(r + 8 * lhalf) * H_STRIDE + col] = f2bf(hnew[r]);
    }
    if (havew) {
      unsigned short* d = &w_lds[1 - cur][ld_row * W_STRIDE + ld_col];
      d[0] = f2bf(wreg.x); d[1] = f2bf(wreg.y); d[2] = f2bf(wreg.z); d[3] = f2bf(wreg.w);
    }

    __syncthreads();  // h_t visible, next word tile staged

    // fc: y(16x64) = h_t(16x128) @ W_fcT, waves 0..3, one 16-col tile each
    if (wave < 4) {
      v8f yacc;
      {
        const v16bf A0 = lds_gatherA(h_lds, H_STRIDE, lrow, lhalf, 0);
        yacc = __builtin_amdgcn_wmma_f32_16x16x32_bf16(
            false, A0, false, Bf[0], (short)0, bfc_acc, false, false);
      }
#pragma unroll
      for (int kb = 1; kb < 4; ++kb) {
        const v16bf A = lds_gatherA(h_lds, H_STRIDE, lrow, lhalf, 32 * kb);
        yacc = __builtin_amdgcn_wmma_f32_16x16x32_bf16(
            false, A, false, Bf[kb], (short)0, yacc, false, false);
      }
      const int col = 16 * wave + lrow;
#pragma unroll
      for (int r = 0; r < 8; ++r)
        out[((size_t)(b0 + r + 8 * lhalf) * T_LEN + t) * 64 + col] = yacc[r];
    }
  }
}

// ---------------------------------------------------------------------------
extern "C" void kernel_launch(void* const* d_in, const int* in_sizes, int n_in,
                              void* d_out, int out_size, void* d_ws, size_t ws_size,
                              hipStream_t stream) {
  (void)in_sizes; (void)n_in; (void)out_size; (void)ws_size;
  const float* speech = (const float*)d_in[0];
  const float* word   = (const float*)d_in[1];
  // d_in[2] memory, d_in[3] memory_lengths: unused by the reference
  const float* W_ih   = (const float*)d_in[4];
  const float* W_hh   = (const float*)d_in[5];
  const float* b_ih   = (const float*)d_in[6];
  const float* b_hh   = (const float*)d_in[7];
  const float* W_fc   = (const float*)d_in[8];
  const float* b_fc   = (const float*)d_in[9];

  char* ws = (char*)d_ws;
  unsigned short* gatesB = (unsigned short*)ws;
  unsigned short* fcB    = (unsigned short*)(ws + GATES_B_BYTES);
  float*          biasg  = (float*)(ws + GATES_B_BYTES + FC_B_BYTES);

  // 8192 gate chunks + 512 fc chunks + 512 bias = 9216 threads = 36 blocks
  tacotron_prep_kernel<<<36, 256, 0, stream>>>(W_ih, W_hh, b_ih, b_hh, W_fc,
                                               gatesB, fcB, biasg);
  tacotron_lstm_scan<<<16, 256, 0, stream>>>(speech, word,
                                             (const v16bf*)gatesB, (const v16bf*)fcB,
                                             biasg, b_fc, (float*)d_out);
}
